// AiriaSNN_2216203125284
// MI455X (gfx1250) — matrix operand order repaired and checked by reading, not compiled, not run on hardware
//
#include <hip/hip_runtime.h>

// ---------------------------------------------------------------------------
// SNN (3-layer LIF, snnTorch Leaky, subtract reset) on MI455X / gfx1250.
//
// Roofline: output = 25*B*3*4B = 315 MB of pure streaming stores ->
// ~13.5 us floor @ 23.3 TB/s. Per-step GEMMs run on v_wmma_f32_16x16x32_f16
// with a BLOCK-DIAGONAL A operand: A = diag(W, W) (rows 0-7 = W on K 0-15,
// rows 8-15 = W on K 16-31). Since this code's B layout (contiguous K) maps lanes 0-15 to
// K 0-15 (N = lane) and lanes 16-31 to K 16-31 (N = lane-16), every lane
// feeding its OWN spikes into B makes one WMMA process all 32 batch
// elements of the wave, and the D result lands back in the full-wave
// per-lane layout (VGPR r = neuron r, lane = batch element). No cross-lane
// data movement anywhere in the time loop; biases fold into the C operand.
// ---------------------------------------------------------------------------

typedef __attribute__((ext_vector_type(16))) _Float16 v16h;
typedef __attribute__((ext_vector_type(8)))  float    v8f;
typedef __attribute__((ext_vector_type(2)))  float    v2f;

#define BETA 0.95f
#define THR  1.0f

__global__ __launch_bounds__(256) void AiriaSNN_kernel(
    const float* __restrict__ x,
    const float* __restrict__ W1, const float* __restrict__ b1,
    const float* __restrict__ W2, const float* __restrict__ b2,
    const float* __restrict__ W3, const float* __restrict__ b3,
    const int*   __restrict__ num_steps_p,
    float* __restrict__ out, int n)
{
    const int gtid = blockIdx.x * blockDim.x + threadIdx.x;  // batch element
    const int lane = threadIdx.x & 31;
    const int T    = *num_steps_p;            // device scalar (graph-safe)
    const int gi   = gtid < n ? gtid : n - 1; // clamp loads; keep EXEC full

    // ---- layer-1 current (fixed across steps): cur1 = x @ W1^T + b1, f32 ---
    float xv[6];
    #pragma unroll
    for (int k = 0; k < 6; ++k)
        xv[k] = __builtin_nontemporal_load(x + (size_t)gi * 6 + k);
    float cur1[16];
    #pragma unroll
    for (int j = 0; j < 16; ++j) {
        float a = b1[j];                       // uniform -> s_load
        #pragma unroll
        for (int k = 0; k < 6; ++k) a = fmaf(W1[j * 6 + k], xv[k], a);
        cur1[j] = a;
    }

    // ---- block-diagonal weights in WMMA A layout ---------------------------
    // A-layout per lane (row = lane&15, kb = (lane>>4)*8):
    //   elems 0..15 = K (2*kb + e)  [lanes 0-15: K 0..15, lanes 16-31: K 16..31]
    // A2 = diag(W2[8x16], W2[8x16]); A3 = diag(W3[3x8], W3[3x8]) zero-padded.
    const int row = lane & 15;
    const int kb  = (lane >> 4) * 8;
    v16h A2, A3;
    #pragma unroll
    for (int e = 0; e < 16; ++e) {
        const int kk = e;                      // K index within a 16-block
        _Float16 w2 = (_Float16)0.f, w3 = (_Float16)0.f;
        if (kb == 0) {                         // diag block 0: rows 0..7/0..2
            if (row < 8)                        w2 = (_Float16)W2[row * 16 + kk];
            if (row < 3 && kk < 8)              w3 = (_Float16)W3[row * 8 + kk];
        } else {                               // diag block 1: rows 8..15
            if (row >= 8)                       w2 = (_Float16)W2[(row - 8) * 16 + kk];
            if (row >= 8 && row < 11 && kk < 8) w3 = (_Float16)W3[(row - 8) * 8 + kk];
        }
        A2[e] = w2; A3[e] = w3;
    }

    // ---- biases as C operand (lanes 16-31 are M=r+8 = tile1 neuron r) ------
    v8f C2, C3;
    #pragma unroll
    for (int r = 0; r < 8; ++r) {
        C2[r] = b2[r];                         // broadcast, valid for both tiles
        C3[r] = (r < 3) ? b3[r] : 0.f;
    }

    // ---- LIF state, full-wave per-lane (prev spikes = detached reset) ------
    float m1[16], s1[16];
    #pragma unroll
    for (int i = 0; i < 16; ++i) { m1[i] = 0.f; s1[i] = 0.f; }
    float m2[8], s2[8];
    #pragma unroll
    for (int r = 0; r < 8; ++r)  { m2[r] = 0.f; s2[r] = 0.f; }
    float m3[3], s3[3];
    #pragma unroll
    for (int c = 0; c < 3; ++c)  { m3[c] = 0.f; s3[c] = 0.f; }

    for (int t = 0; t < T; ++t) {
        // ---- layer 1 leaky; every lane packs its OWN spikes into B ---------
        v16h B1;
        #pragma unroll
        for (int i = 0; i < 16; ++i) {
            m1[i] = fmaf(BETA, m1[i], cur1[i]) - s1[i] * THR;
            s1[i] = (m1[i] > THR) ? 1.f : 0.f;
            B1[i] = (_Float16)s1[i];
        }

        // ---- layer 2: one WMMA covers all 32 batch elements ----------------
        v8f d2 = __builtin_amdgcn_wmma_f32_16x16x32_f16(
            false, A2, false, B1, (short)0, C2, false, false);

        // ---- layer 2 leaky (full wave); repack spikes lane-locally ---------
        v16h B2;
        #pragma unroll
        for (int r = 0; r < 8; ++r) {
            m2[r] = fmaf(BETA, m2[r], d2[r]) - s2[r] * THR;
            s2[r] = (m2[r] > THR) ? 1.f : 0.f;
            B2[r]     = (_Float16)s2[r];
            B2[r + 8] = (_Float16)0.f;         // K 8..15 / 24..31 unused (K=8)
        }

        // ---- layer 3: one WMMA -------------------------------------------
        v8f d3 = __builtin_amdgcn_wmma_f32_16x16x32_f16(
            false, A3, false, B2, (short)0, C3, false, false);

        // ---- layer 3 leaky ------------------------------------------------
        #pragma unroll
        for (int c = 0; c < 3; ++c) {
            m3[c] = fmaf(BETA, m3[c], d3[c]) - s3[c] * THR;
            s3[c] = (m3[c] > THR) ? 1.f : 0.f;
        }

        // ---- single guarded block: 12B/lane contiguous NT stores -----------
        if (gtid < n) {                        // one EXEC toggle per step
            float* op = out + (size_t)t * (size_t)n * 3 + (size_t)gtid * 3;
            v2f lo; lo[0] = s3[0]; lo[1] = s3[1];
            __builtin_nontemporal_store(lo, (v2f*)op);          // b64
            __builtin_nontemporal_store(s3[2], op + 2);         // b32
        }                                      // EXEC restored before next WMMA
    }
}

extern "C" void kernel_launch(void* const* d_in, const int* in_sizes, int n_in,
                              void* d_out, int out_size, void* d_ws, size_t ws_size,
                              hipStream_t stream) {
    const float* x  = (const float*)d_in[0];
    const float* W1 = (const float*)d_in[1];
    const float* b1 = (const float*)d_in[2];
    const float* W2 = (const float*)d_in[3];
    const float* b2 = (const float*)d_in[4];
    const float* W3 = (const float*)d_in[5];
    const float* b3 = (const float*)d_in[6];
    const int* num_steps = (const int*)d_in[7];
    const int n = in_sizes[0] / 6;             // batch size (1,048,576)

    const int block = 256;                      // 8 wave32 per block
    const int grid  = (n + block - 1) / block;
    AiriaSNN_kernel<<<grid, block, 0, stream>>>(
        x, W1, b1, W2, b2, W3, b3, num_steps, (float*)d_out, n);
}
